// LogisticRegression_63694365000268
// MI455X (gfx1250) — matrix operand order repaired and checked by reading, not compiled, not run on hardware
//
#include <hip/hip_runtime.h>
#include <hip/hip_bf16.h>
#include <math.h>

#define VOCAB   50000
#define EMB     300
#define BATCH   1024
#define SEQ     200
#define BM_WORDS 1563   // ceil(50000/32)

typedef __attribute__((ext_vector_type(16))) _Float16 v16h;
typedef __attribute__((ext_vector_type(8)))  float    v8f;

// ---------------------------------------------------------------------------
// Kernel 1: dot[t] = <emb_table[t, 0:300], W[0:300]>  via WMMA f16 -> f32.
// One wave32 per 16 table rows.  D = A(16x32) x B(32x16) + C with
//   A[m][k] = W[kb+k]            (same for every m -> all D rows identical)
//   B[k][n] = emb[rowBase+n][kb+k]
// A f16 layout (same K order as B): lanes 0-15 hold K 0..15, lanes 16-31
// hold K 16..31.  B f16 layout: lane L holds N=L&15, K = (L>>4)*16+j
// for element j -> 16 consecutive floats per lane.
// ---------------------------------------------------------------------------
__global__ __launch_bounds__(32)
void tok_dot_kernel(const float* __restrict__ emb,
                    const float* __restrict__ W,
                    float* __restrict__ dot)
{
    const int lane    = threadIdx.x;        // 0..31
    const int half    = lane >> 4;          // 0 or 1
    const int col     = lane & 15;          // N / table row within tile
    const int rowBase = blockIdx.x * 16;    // 3125 blocks * 16 = 50000 exactly
    const int rB      = (rowBase + col) * EMB;

    v8f c = {};

    for (int kb = 0; kb < EMB; kb += 32) {
        v16h a, b;
        const int aoff = kb + half * 16;    // A: this half-wave's K window
        const int boff = kb + half * 16;    // B: this half-wave's K window

        if (kb + 32 <= EMB) {
            // ---- full step: 16B-aligned vector loads ----
            const float4* wa = (const float4*)(W + aoff);
            float4 w0 = wa[0], w1 = wa[1];                 // W[aoff .. aoff+7]
            const float4* wb = (const float4*)(W + aoff + 8);
            float4 w2 = wb[0], w3 = wb[1];                 // W[aoff+8 .. aoff+15]
            a[0]=(_Float16)w0.x;  a[1]=(_Float16)w0.y;  a[2]=(_Float16)w0.z;  a[3]=(_Float16)w0.w;
            a[4]=(_Float16)w1.x;  a[5]=(_Float16)w1.y;  a[6]=(_Float16)w1.z;  a[7]=(_Float16)w1.w;
            a[8]=(_Float16)w2.x;  a[9]=(_Float16)w2.y;  a[10]=(_Float16)w2.z; a[11]=(_Float16)w2.w;
            a[12]=(_Float16)w3.x; a[13]=(_Float16)w3.y; a[14]=(_Float16)w3.z; a[15]=(_Float16)w3.w;

            const float4* bp = (const float4*)(emb + rB + boff);
            float4 e0 = bp[0], e1 = bp[1], e2 = bp[2], e3 = bp[3];
            b[0]=(_Float16)e0.x;  b[1]=(_Float16)e0.y;  b[2]=(_Float16)e0.z;  b[3]=(_Float16)e0.w;
            b[4]=(_Float16)e1.x;  b[5]=(_Float16)e1.y;  b[6]=(_Float16)e1.z;  b[7]=(_Float16)e1.w;
            b[8]=(_Float16)e2.x;  b[9]=(_Float16)e2.y;  b[10]=(_Float16)e2.z; b[11]=(_Float16)e2.w;
            b[12]=(_Float16)e3.x; b[13]=(_Float16)e3.y; b[14]=(_Float16)e3.z; b[15]=(_Float16)e3.w;
        } else {
            // ---- tail step (kb = 288): zero-pad K 300..319 ----
            #pragma unroll
            for (int j = 0; j < 8; ++j) {
                int k0 = aoff + j, k1 = aoff + 8 + j;
                a[j]     = (_Float16)((k0 < EMB) ? W[k0] : 0.0f);
                a[8 + j] = (_Float16)((k1 < EMB) ? W[k1] : 0.0f);
            }
            #pragma unroll
            for (int j = 0; j < 16; ++j) {
                int k = boff + j;
                b[j] = (_Float16)((k < EMB) ? emb[rB + k] : 0.0f);
            }
        }

        // (neg_a, A, neg_b, B, c_mod, C, reuse_a, reuse_b)
        c = __builtin_amdgcn_wmma_f32_16x16x32_f16(false, a, false, b,
                                                   (short)0, c, false, false);
    }

    // D row M=0 lives in c[0] of lanes 0..15 (N = lane)
    if (lane < 16) dot[rowBase + col] = c[0];
}

// ---------------------------------------------------------------------------
// Kernel 2: one workgroup (256 thr) per batch row.
//   logit = (1/S) * sum_j dot[x[j]]  +  sum_{distinct t} W[300+t]  +  b
// Dedup via LDS vocab bitmap + ds_or_rtn_b32 (exactly one winner per token).
// ---------------------------------------------------------------------------
__global__ __launch_bounds__(256)
void row_logit_kernel(const int*   __restrict__ x,
                      const float* __restrict__ dot,
                      const float* __restrict__ W,
                      const float* __restrict__ bias,
                      float*       __restrict__ out)
{
    __shared__ unsigned bitmap[BM_WORDS];
    __shared__ float    red[256];

    const int row = blockIdx.x;
    const int tid = threadIdx.x;

    for (int w = tid; w < BM_WORDS; w += 256) bitmap[w] = 0u;
    __syncthreads();

    float contrib = 0.0f;
    if (tid < SEQ) {
        int t = x[row * SEQ + tid];
        unsigned bit = 1u << (t & 31);
        unsigned old = atomicOr(&bitmap[t >> 5], bit);   // ds_or_rtn_b32
        contrib = dot[t] * (1.0f / (float)SEQ);          // duplicates count in mean
        if ((old & bit) == 0u)                           // first sighting only
            contrib += W[EMB + t];
    }
    red[tid] = contrib;
    __syncthreads();

    #pragma unroll
    for (int s = 128; s > 0; s >>= 1) {
        if (tid < s) red[tid] += red[tid + s];
        __syncthreads();
    }

    if (tid == 0) {
        float z = red[0] + bias[0];
        out[row] = 1.0f / (1.0f + expf(-z));
    }
}

// ---------------------------------------------------------------------------
extern "C" void kernel_launch(void* const* d_in, const int* in_sizes, int n_in,
                              void* d_out, int out_size, void* d_ws, size_t ws_size,
                              hipStream_t stream)
{
    const int*   x   = (const int*)  d_in[0];   // [1024, 200] int32
    const float* emb = (const float*)d_in[1];   // [50000, 300] f32
    const float* W   = (const float*)d_in[2];   // [1, 50300]  f32
    const float* b   = (const float*)d_in[3];   // [1]         f32
    float* out = (float*)d_out;                 // [1024]      f32

    float* dot = (float*)d_ws;                  // 50000 floats of scratch

    // Phase 1: token dot-products via WMMA (3125 waves cover 50000 rows)
    tok_dot_kernel<<<VOCAB / 16, 32, 0, stream>>>(emb, W, dot);

    // Phase 2: per-row dedup + reduction + sigmoid
    row_logit_kernel<<<BATCH, 256, 0, stream>>>(x, dot, W, b, out);
}
